// MambaModel_54795192762436
// MI455X (gfx1250) — matrix operand order repaired and checked by reading, not compiled, not run on hardware
//
#include <hip/hip_runtime.h>
#include <hip/hip_bf16.h>

typedef __attribute__((ext_vector_type(16))) _Float16 v16h;
typedef __attribute__((ext_vector_type(8)))  _Float16 v8h;
typedef __attribute__((ext_vector_type(8)))  float    v8f;

constexpr int kB    = 2;
constexpr int kL    = 4096;
constexpr int kDM   = 256;
constexpr int kDOUT = 128;
constexpr int kDST  = 16;
constexpr int kDC   = 3;
constexpr int kDI   = 512;          // EXP * DM
constexpr int kDTR  = 16;           // (DM+15)/16
constexpr int kBL   = kB * kL;      // 8192 rows
constexpr int kNXZ  = 2 * kDI;      // 1024
constexpr int kNX   = kDTR + 2 * kDST; // 48
constexpr int kNC   = 64;           // scan chunks
constexpr int kCHL  = kL / kNC;     // 64 steps per chunk

// ---------------------------------------------------------------------------
// Generic f32 -> f16 conversion (weights, done once per launch)
// ---------------------------------------------------------------------------
__global__ void cvt_f32_f16(const float* __restrict__ s,
                            _Float16* __restrict__ d, int n) {
  int i = blockIdx.x * blockDim.x + threadIdx.x;
  if (i < n) d[i] = (_Float16)s[i];
}

// ---------------------------------------------------------------------------
// WMMA GEMM: C[M,N] = A[M,K] (f16, row-major) x W[N,K]^T (f16, row-major)
// One wave per 16x16 output tile; 4 waves per block cover 4 N-tiles.
// A layout: lane=row, K = kb + hi*16 + e (same order as B)
// => two contiguous 8-half (b128) loads.
// B layout: lane=col, K = kb + hi*16 + e => 16 contiguous halfs along K.
// ---------------------------------------------------------------------------
__global__ void gemm_f16_wmma(const _Float16* __restrict__ A,
                              const _Float16* __restrict__ W,
                              float* __restrict__ C,
                              int M, int N, int K) {
  int lane  = threadIdx.x;            // 0..31
  int wv    = threadIdx.y;            // 0..3
  int ntile = blockIdx.x * 4 + wv;
  if (ntile * 16 >= N) return;
  int mtile = blockIdx.y;
  int lo = lane & 15;
  int hi = lane >> 4;

  const _Float16* ap = A + (size_t)(mtile * 16 + lo) * K + hi * 16;
  const _Float16* bp = W + (size_t)(ntile * 16 + lo) * K + hi * 16;

  v8f acc = {0.f, 0.f, 0.f, 0.f, 0.f, 0.f, 0.f, 0.f};
  for (int kb = 0; kb < K; kb += 32) {
    v8h alo = *(const v8h*)(ap + kb);
    v8h ahi = *(const v8h*)(ap + kb + 8);
    v16h a, b;
#pragma unroll
    for (int i = 0; i < 8; ++i) { a[i] = alo[i]; a[i + 8] = ahi[i]; }
    b = *(const v16h*)(bp + kb);
    __builtin_prefetch(bp + kb + 64, 0, 1);   // global_prefetch_b8
    acc = __builtin_amdgcn_wmma_f32_16x16x32_f16(
        /*neg_a=*/false, a, /*neg_b=*/false, b,
        /*c_mod=*/(short)0, acc, /*reuse_a=*/false, /*reuse_b=*/false);
  }
  // D layout: VGPR r -> M = mtile*16 + r + 8*hi, N = ntile*16 + (lane&15)
  float* cp = C + (size_t)(mtile * 16 + 8 * hi) * N + ntile * 16 + lo;
#pragma unroll
  for (int r = 0; r < 8; ++r) cp[(size_t)r * N] = acc[r];
}

// ---------------------------------------------------------------------------
// Residual add + LayerNorm (DM=256, one row per block), f16 output for GEMM
// ---------------------------------------------------------------------------
__global__ void ln_res_kernel(const float* __restrict__ src,
                              float* __restrict__ res, int add,
                              const float* __restrict__ w,
                              const float* __restrict__ bias,
                              _Float16* __restrict__ out_h) {
  __shared__ float s1[kDM], s2[kDM];
  int t = threadIdx.x;
  size_t i = (size_t)blockIdx.x * kDM + t;
  float r = src[i];
  if (add) r += res[i];
  res[i] = r;
  s1[t] = r; s2[t] = r * r;
  __syncthreads();
  for (int s = kDM / 2; s > 0; s >>= 1) {
    if (t < s) { s1[t] += s1[t + s]; s2[t] += s2[t + s]; }
    __syncthreads();
  }
  float mean = s1[0] * (1.f / kDM);
  float var  = s2[0] * (1.f / kDM) - mean * mean;
  float inv  = rsqrtf(var + 1e-5f);
  out_h[i] = (_Float16)((r - mean) * inv * w[t] + bias[t]);
}

// ---------------------------------------------------------------------------
// Depthwise causal conv (DC=3) + bias + SiLU; reads u half of xz
// ---------------------------------------------------------------------------
__global__ void conv_silu_kernel(const float* __restrict__ xz,
                                 const float* __restrict__ cw,
                                 const float* __restrict__ cb,
                                 float* __restrict__ uc,
                                 _Float16* __restrict__ u_h) {
  size_t idx = (size_t)blockIdx.x * blockDim.x + threadIdx.x;
  if (idx >= (size_t)kBL * kDI) return;
  int d = (int)(idx % kDI);
  size_t bl = idx / kDI;
  int l = (int)(bl % kL);
  size_t brow = bl - l;  // b*L
  float acc = cb[d];
#pragma unroll
  for (int j = 0; j < kDC; ++j) {
    int lp = l - (kDC - 1) + j;
    if (lp >= 0) acc += cw[d * kDC + j] * xz[(brow + lp) * kNXZ + d];
  }
  float s = acc / (1.f + __expf(-acc));   // silu
  uc[idx]  = s;
  u_h[idx] = (_Float16)s;
}

// ---------------------------------------------------------------------------
// dt = softplus(xdbl[:, :16] @ dtw^T + dtb)  (K=16 -> VALU, row in LDS)
// ---------------------------------------------------------------------------
__global__ void dtproj_kernel(const float* __restrict__ xdbl,
                              const float* __restrict__ dtw,
                              const float* __restrict__ dtb,
                              float* __restrict__ dtf) {
  __shared__ float sx[kDTR];
  int t = threadIdx.x;
  size_t bl = blockIdx.x;
  if (t < kDTR) sx[t] = xdbl[bl * kNX + t];
  __syncthreads();
  for (int d = t; d < kDI; d += blockDim.x) {
    float acc = dtb[d];
#pragma unroll
    for (int r = 0; r < kDTR; ++r) acc += sx[r] * dtw[d * kDTR + r];
    dtf[bl * kDI + d] = (acc > 20.f) ? acc : log1pf(__expf(acc));
  }
}

// ---------------------------------------------------------------------------
// Chunked selective scan, pass 1: per (b, chunk, d) local scan with h0=0.
// Emits chunk end-state S and sum(dt) (chunk decay = exp(A * sum dt)).
// ---------------------------------------------------------------------------
__global__ void scan_pass1(const float* __restrict__ xdbl,
                           const float* __restrict__ dtf,
                           const float* __restrict__ uc,
                           const float* __restrict__ A_log,
                           float* __restrict__ Sbuf,
                           float* __restrict__ dtsum) {
  __shared__ float sB[kCHL][kDST];
  int t = threadIdx.x;                  // 0..127
  int d = blockIdx.x * 128 + t;
  int c = blockIdx.y;
  int b = blockIdx.z;
  int l0 = c * kCHL;
  for (int i = t; i < kCHL * kDST; i += 128) {
    int ll = i >> 4, n = i & 15;
    sB[ll][n] = xdbl[((size_t)b * kL + l0 + ll) * kNX + kDTR + n];
  }
  __syncthreads();
  float A[kDST], h[kDST];
#pragma unroll
  for (int n = 0; n < kDST; ++n) { A[n] = -__expf(A_log[d * kDST + n]); h[n] = 0.f; }
  float ds = 0.f;
  for (int lc = 0; lc < kCHL; ++lc) {
    size_t bl = (size_t)b * kL + l0 + lc;
    float dtv = dtf[bl * kDI + d];
    float uv  = uc[bl * kDI + d];
    ds += dtv;
    float x = dtv * uv;
#pragma unroll
    for (int n = 0; n < kDST; ++n)
      h[n] = __expf(dtv * A[n]) * h[n] + x * sB[lc][n];
  }
  size_t idx = (size_t)(b * kNC + c) * kDI + d;
#pragma unroll
  for (int n = 0; n < kDST; ++n) Sbuf[idx * kDST + n] = h[n];
  dtsum[idx] = ds;
}

// ---------------------------------------------------------------------------
// Carry pass: sequentially combine the kNC chunk summaries per channel.
// Hbuf[c] = state entering chunk c.
// ---------------------------------------------------------------------------
__global__ void scan_carry(const float* __restrict__ Sbuf,
                           const float* __restrict__ dtsum,
                           const float* __restrict__ A_log,
                           float* __restrict__ Hbuf) {
  int t = threadIdx.x;
  int d = blockIdx.x * 256 + t;
  int b = blockIdx.z;
  float A[kDST], h[kDST];
#pragma unroll
  for (int n = 0; n < kDST; ++n) { A[n] = -__expf(A_log[d * kDST + n]); h[n] = 0.f; }
  for (int c = 0; c < kNC; ++c) {
    size_t idx = (size_t)(b * kNC + c) * kDI + d;
#pragma unroll
    for (int n = 0; n < kDST; ++n) Hbuf[idx * kDST + n] = h[n];
    float p = dtsum[idx];
#pragma unroll
    for (int n = 0; n < kDST; ++n)
      h[n] = __expf(p * A[n]) * h[n] + Sbuf[idx * kDST + n];
  }
}

// ---------------------------------------------------------------------------
// Pass 2: re-run each chunk from corrected initial state; emit
// y = (scan_out + u*D) * silu(z) as f16 for the output GEMM.
// ---------------------------------------------------------------------------
__global__ void scan_pass2(const float* __restrict__ xdbl,
                           const float* __restrict__ dtf,
                           const float* __restrict__ uc,
                           const float* __restrict__ xz,
                           const float* __restrict__ A_log,
                           const float* __restrict__ Dv,
                           const float* __restrict__ Hbuf,
                           _Float16* __restrict__ y_h) {
  __shared__ float sBC[kCHL][2 * kDST];
  int t = threadIdx.x;                  // 0..127
  int d = blockIdx.x * 128 + t;
  int c = blockIdx.y;
  int b = blockIdx.z;
  int l0 = c * kCHL;
  for (int i = t; i < kCHL * 2 * kDST; i += 128) {
    int ll = i >> 5, n = i & 31;
    sBC[ll][n] = xdbl[((size_t)b * kL + l0 + ll) * kNX + kDTR + n];
  }
  __syncthreads();
  float A[kDST], h[kDST];
  size_t idx = (size_t)(b * kNC + c) * kDI + d;
#pragma unroll
  for (int n = 0; n < kDST; ++n) {
    A[n] = -__expf(A_log[d * kDST + n]);
    h[n] = Hbuf[idx * kDST + n];
  }
  float Dd = Dv[d];
  for (int lc = 0; lc < kCHL; ++lc) {
    size_t bl = (size_t)b * kL + l0 + lc;
    float dtv = dtf[bl * kDI + d];
    float uv  = uc[bl * kDI + d];
    float zv  = xz[bl * kNXZ + kDI + d];
    float x = dtv * uv;
    float y = 0.f;
#pragma unroll
    for (int n = 0; n < kDST; ++n) {
      h[n] = __expf(dtv * A[n]) * h[n] + x * sBC[lc][n];
      y += h[n] * sBC[lc][kDST + n];
    }
    y = (y + uv * Dd) * (zv / (1.f + __expf(-zv)));
    y_h[bl * kDI + d] = (_Float16)y;
  }
}

// ---------------------------------------------------------------------------
extern "C" void kernel_launch(void* const* d_in, const int* in_sizes, int n_in,
                              void* d_out, int out_size, void* d_ws, size_t ws_size,
                              hipStream_t stream) {
  const float* x        = (const float*)d_in[0];
  const float* midW_in  = (const float*)d_in[1];
  const float* midW_cw  = (const float*)d_in[2];
  const float* midW_cb  = (const float*)d_in[3];
  const float* midW_x   = (const float*)d_in[4];
  const float* midW_dtw = (const float*)d_in[5];
  const float* midW_dtb = (const float*)d_in[6];
  const float* midA_log = (const float*)d_in[7];
  const float* midD     = (const float*)d_in[8];
  const float* midW_out = (const float*)d_in[9];
  const float* midLNw   = (const float*)d_in[10];
  const float* midLNb   = (const float*)d_in[11];
  const float* finW_in  = (const float*)d_in[12];
  const float* finW_cw  = (const float*)d_in[13];
  const float* finW_cb  = (const float*)d_in[14];
  const float* finW_x   = (const float*)d_in[15];
  const float* finW_dtw = (const float*)d_in[16];
  const float* finW_dtb = (const float*)d_in[17];
  const float* finA_log = (const float*)d_in[18];
  const float* finD     = (const float*)d_in[19];
  const float* finW_out = (const float*)d_in[20];
  const float* finLNw   = (const float*)d_in[21];
  const float* finLNb   = (const float*)d_in[22];

  char* p = (char*)d_ws;
  auto carve = [&p](size_t bytes) -> char* {
    char* r = p;
    p += (bytes + 255) & ~(size_t)255;
    return r;
  };
  float*    res   = (float*)   carve((size_t)kBL * kDM * 4);
  float*    hbuf  = (float*)   carve((size_t)kBL * kDM * 4);
  _Float16* hn_h  = (_Float16*)carve((size_t)kBL * kDM * 2);
  float*    xz    = (float*)   carve((size_t)kBL * kNXZ * 4);
  float*    ucb   = (float*)   carve((size_t)kBL * kDI * 4);
  _Float16* u_h   = (_Float16*)carve((size_t)kBL * kDI * 2);
  float*    xdbl  = (float*)   carve((size_t)kBL * kNX * 4);
  float*    dtf   = (float*)   carve((size_t)kBL * kDI * 4);
  _Float16* y_h   = (_Float16*)carve((size_t)kBL * kDI * 2);
  float*    Sbuf  = (float*)   carve((size_t)kB * kNC * kDI * kDST * 4);
  float*    Hbuf  = (float*)   carve((size_t)kB * kNC * kDI * kDST * 4);
  float*    dtsum = (float*)   carve((size_t)kB * kNC * kDI * 4);
  _Float16* wInH  = (_Float16*)carve((size_t)3 * kNXZ * kDM * 2);
  _Float16* wXH   = (_Float16*)carve((size_t)3 * kNX * kDI * 2);
  _Float16* wOutH = (_Float16*)carve((size_t)(2 * kDM + kDOUT) * kDI * 2);

  auto cvt = [&](const float* s, _Float16* d, int n) {
    cvt_f32_f16<<<(n + 255) / 256, 256, 0, stream>>>(s, d, n);
  };
  cvt(midW_in,  wInH,                    2 * kNXZ * kDM);
  cvt(finW_in,  wInH + 2 * kNXZ * kDM,       kNXZ * kDM);
  cvt(midW_x,   wXH,                     2 * kNX * kDI);
  cvt(finW_x,   wXH + 2 * kNX * kDI,         kNX * kDI);
  cvt(midW_out, wOutH,                   2 * kDM * kDI);
  cvt(finW_out, wOutH + 2 * kDM * kDI,       kDOUT * kDI);

  unsigned int convBlocks = (unsigned int)(((size_t)kBL * kDI + 255) / 256);

  for (int blk = 0; blk < 3; ++blk) {
    bool fin = (blk == 2);
    const float*    LNw  = fin ? finLNw   : midLNw  + blk * kDM;
    const float*    LNb  = fin ? finLNb   : midLNb  + blk * kDM;
    const _Float16* wIn  = wInH + (fin ? 2 : blk) * kNXZ * kDM;
    const float*    cw   = fin ? finW_cw  : midW_cw + blk * kDI * kDC;
    const float*    cb   = fin ? finW_cb  : midW_cb + blk * kDI;
    const _Float16* wX   = wXH + (fin ? 2 : blk) * kNX * kDI;
    const float*    dtw  = fin ? finW_dtw : midW_dtw + blk * kDI * kDTR;
    const float*    dtb  = fin ? finW_dtb : midW_dtb + blk * kDI;
    const float*    Alog = fin ? finA_log : midA_log + blk * kDI * kDST;
    const float*    Dv   = fin ? finD     : midD     + blk * kDI;
    const _Float16* wOut = fin ? (wOutH + 2 * kDM * kDI) : (wOutH + blk * kDM * kDI);
    int    Nout = fin ? kDOUT : kDM;
    float* outC = fin ? (float*)d_out : hbuf;

    // 1. residual accumulate + layernorm -> hn (f16)
    ln_res_kernel<<<kBL, kDM, 0, stream>>>(blk == 0 ? x : hbuf, res,
                                           blk != 0 ? 1 : 0, LNw, LNb, hn_h);
    // 2. xz = hn @ W_in^T   (8192 x 1024, K=256)
    gemm_f16_wmma<<<dim3((kNXZ / 16 + 3) / 4, kBL / 16), dim3(32, 4), 0, stream>>>(
        hn_h, wIn, xz, kBL, kNXZ, kDM);
    // 3. depthwise causal conv + SiLU -> uc (f32) + u_h (f16)
    conv_silu_kernel<<<convBlocks, 256, 0, stream>>>(xz, cw, cb, ucb, u_h);
    // 4. xdbl = u @ W_x^T   (8192 x 48, K=512)
    gemm_f16_wmma<<<dim3((kNX / 16 + 3) / 4, kBL / 16), dim3(32, 4), 0, stream>>>(
        u_h, wX, xdbl, kBL, kNX, kDI);
    // 5. dt = softplus(dt_part @ dtw^T + dtb)
    dtproj_kernel<<<kBL, 256, 0, stream>>>(xdbl, dtw, dtb, dtf);
    // 6. chunked parallel selective scan (3 phases)
    scan_pass1<<<dim3(kDI / 128, kNC, kB), 128, 0, stream>>>(
        xdbl, dtf, ucb, Alog, Sbuf, dtsum);
    scan_carry<<<dim3(kDI / 256, 1, kB), 256, 0, stream>>>(
        Sbuf, dtsum, Alog, Hbuf);
    scan_pass2<<<dim3(kDI / 128, kNC, kB), 128, 0, stream>>>(
        xdbl, dtf, ucb, xz, Alog, Dv, Hbuf, y_h);
    // 7. out = y @ W_out^T  (mid: -> hbuf, fin: -> d_out)
    gemm_f16_wmma<<<dim3((Nout / 16 + 3) / 4, kBL / 16), dim3(32, 4), 0, stream>>>(
        y_h, wOut, outC, kBL, Nout, kDI);
  }
}
